// LocalMatching_2637109920016
// MI455X (gfx1250) — matrix operand order repaired and checked by reading, not compiled, not run on hardware
//
#include <hip/hip_runtime.h>

// ---------------------------------------------------------------------------
// ESIM-style local matching, fused flash-attention + matching-FC for gfx1250.
//   cl = relu(concat(cs, cd, cs-cd, cs*cd) @ W1^T + b1),  cd = Attn(cs; rs, rs)
//   rl = relu(concat(rs, rd, rs-rd, rs*rd) @ W2^T + b2),  rd = Attn(rs; cs, cs)
// (col-softmax beta == row-softmax of e^T, so both sides are flash attention)
// All GEMMs: v_wmma_f32_16x16x32_bf16 (bf16 in, f32 accumulate).
// ---------------------------------------------------------------------------

typedef __attribute__((ext_vector_type(16))) __bf16 v16bf;
typedef __attribute__((ext_vector_type(8)))  __bf16 v8bf;
typedef __attribute__((ext_vector_type(8)))  float  v8f;
typedef __attribute__((ext_vector_type(4)))  float  vf4;

#define DIM   256      // D
#define HID   128      // H
#define FDIM  1024     // 4*D
#define KT    32       // keys per streamed tile

// LDS strides (bf16 elements), padded so 16-lane row accesses hit distinct banks
#define KVN_S 264      // row-major KV tile:   32 rows x 264
#define KVT_S 40       // transposed KV tile: 256 rows x 40
#define PP_S  40       // per-wave P tile:     16 rows x 40
#define OB_S  264      // per-wave O staging:  16 rows x 264 (bf16)

#define KVN_BYTES (32  * KVN_S * 2)                 // 16896
#define KVT_BYTES (256 * KVT_S * 2)                 // 20480
#define PP_BYTES  (16  * PP_S  * 2)                 // 1280 per wave
#define SMEM_BYTES (KVN_BYTES + KVT_BYTES + 4 * PP_BYTES)   // 42496 (< 320KB WGP LDS)

static __device__ __forceinline__ v8f wmma_bf16(v16bf a, v16bf b, v8f c) {
  // (neg_a, A, neg_b, B, c_mod, C, reuse_a, reuse_b)
  return __builtin_amdgcn_wmma_f32_16x16x32_bf16(false, a, false, b, (short)0, c,
                                                 false, false);
}

// Convert two groups of 8 consecutive f32 into a v16bf A/B fragment register.
static __device__ __forceinline__ v16bf cvt16(const float* p0, const float* p1) {
  vf4 a0 = *(const vf4*)(p0);
  vf4 a1 = *(const vf4*)(p0 + 4);
  vf4 b0 = *(const vf4*)(p1);
  vf4 b1 = *(const vf4*)(p1 + 4);
  v16bf r;
  r[0]  = (__bf16)a0.x; r[1]  = (__bf16)a0.y; r[2]  = (__bf16)a0.z; r[3]  = (__bf16)a0.w;
  r[4]  = (__bf16)a1.x; r[5]  = (__bf16)a1.y; r[6]  = (__bf16)a1.z; r[7]  = (__bf16)a1.w;
  r[8]  = (__bf16)b0.x; r[9]  = (__bf16)b0.y; r[10] = (__bf16)b0.z; r[11] = (__bf16)b0.w;
  r[12] = (__bf16)b1.x; r[13] = (__bf16)b1.y; r[14] = (__bf16)b1.z; r[15] = (__bf16)b1.w;
  return r;
}

// two groups of 8 bf16 from LDS -> B fragment (elems 0..7 = K = 8*half + 0..7, 8..15 = K = 16 + 8*half + 0..7)
static __device__ __forceinline__ v16bf bfrag16(const __bf16* p) {
  v8bf lo = *(const v8bf*)p;
  v8bf hi = *(const v8bf*)(p + 16);
  v16bf r;
#pragma unroll
  for (int i = 0; i < 8; ++i) { r[i] = lo[i]; r[8 + i] = hi[i]; }
  return r;
}

// Row (16-lane half-group) reductions; xor masks < 16 stay inside the half.
static __device__ __forceinline__ float rowmax16(float v) {
#pragma unroll
  for (int m = 1; m < 16; m <<= 1) v = fmaxf(v, __shfl_xor(v, m, 32));
  return v;
}
static __device__ __forceinline__ float rowsum16(float v) {
#pragma unroll
  for (int m = 1; m < 16; m <<= 1) v += __shfl_xor(v, m, 32);
  return v;
}

// ---------------------------------------------------------------------------
// One wave handles 16 query rows; workgroup (4 waves) handles 64 rows and
// cooperatively streams 32-key KV tiles through LDS. After attention, the
// same wave fuses the concat-feature FC (K=1024 -> H=128) + bias + ReLU.
// ---------------------------------------------------------------------------
__global__ void __launch_bounds__(128)
attn_match_fc(const float* __restrict__ q,    // [Lq, D] per batch
              const float* __restrict__ kv,   // [Lk, D] per batch
              const float* __restrict__ W,    // [H, 4D]
              const float* __restrict__ bias, // [H]
              float* __restrict__ out,        // [Lq, H] per batch
              int Lq, int Lk) {
  extern __shared__ char smem[];
  __bf16* kvN = (__bf16*)smem;                              // 32 x KVN_S
  __bf16* kvT = (__bf16*)(smem + KVN_BYTES);                // 256 x KVT_S

  const int tid  = threadIdx.x;
  const int w    = tid >> 5;
  const int lane = tid & 31;
  const int lm   = lane & 15;
  const int half = lane >> 4;

  __bf16* pP = (__bf16*)(smem + KVN_BYTES + KVT_BYTES) + w * 16 * PP_S;
  __bf16* oB = (__bf16*)smem + w * 16 * OB_S;  // reuses kv region after attention

  const int    b     = blockIdx.y;
  const float* qb    = q  + (size_t)b * Lq * DIM;
  const float* kb    = kv + (size_t)b * Lk * DIM;
  float*       ob    = out + (size_t)b * Lq * HID;
  const int    qbase = blockIdx.x * 64 + w * 16;
  const float* qrow  = qb + (size_t)(qbase + lm) * DIM;     // this lane's M row

  // ---- Q fragments (A-matrix layout), built once: 8 chunks of K=32 --------
  v16bf qA[8];
#pragma unroll
  for (int c = 0; c < 8; ++c) {
    const float* p = qrow + 32 * c + 8 * half;
    qA[c] = cvt16(p, p + 16);
  }

  // ---- online-softmax state ----------------------------------------------
  v8f O[16];                       // 16x256 f32 accumulator (C/D layout)
#pragma unroll
  for (int d = 0; d < 16; ++d)
#pragma unroll
    for (int i = 0; i < 8; ++i) O[d][i] = 0.f;
  float M[8], S[8];
#pragma unroll
  for (int i = 0; i < 8; ++i) { M[i] = -1e30f; S[i] = 0.f; }

  const int nkt = Lk / KT;
  for (int kt = 0; kt < nkt; ++kt) {
    // ---- cooperative KV tile staging: f32 global -> bf16 LDS (x2 layouts) --
    {
      const int    row = tid >> 2;
      const int    cb  = (tid & 3) * 64;
      const float* src = kb + (size_t)(kt * KT + row) * DIM + cb;
      if (kt + 1 < nkt)
        __builtin_prefetch(kb + (size_t)((kt + 1) * KT + row) * DIM + cb, 0, 1);
#pragma unroll
      for (int j = 0; j < 64; j += 8) {
        vf4 x0 = *(const vf4*)(src + j);
        vf4 x1 = *(const vf4*)(src + j + 4);
        v8bf h;
        h[0] = (__bf16)x0.x; h[1] = (__bf16)x0.y; h[2] = (__bf16)x0.z; h[3] = (__bf16)x0.w;
        h[4] = (__bf16)x1.x; h[5] = (__bf16)x1.y; h[6] = (__bf16)x1.z; h[7] = (__bf16)x1.w;
        *(v8bf*)&kvN[row * KVN_S + cb + j] = h;
#pragma unroll
        for (int e = 0; e < 8; ++e) kvT[(cb + j + e) * KVT_S + row] = h[e];
      }
    }
    __syncthreads();

    // ---- S = Q * K^T : two 16x16 tiles over this 32-key tile ---------------
    v8f s0 = {0.f, 0.f, 0.f, 0.f, 0.f, 0.f, 0.f, 0.f};
    v8f s1 = s0;
#pragma unroll
    for (int c = 0; c < 8; ++c) {
      s0 = wmma_bf16(qA[c], bfrag16(&kvN[lm * KVN_S + 32 * c + 8 * half]), s0);
      s1 = wmma_bf16(qA[c], bfrag16(&kvN[(16 + lm) * KVN_S + 32 * c + 8 * half]), s1);
    }

    // ---- online softmax (row = i + 8*half for this lane) -------------------
    float corr[8];
#pragma unroll
    for (int i = 0; i < 8; ++i) {
      float rm = rowmax16(fmaxf(s0[i], s1[i]));
      float nM = fmaxf(M[i], rm);
      corr[i]  = __expf(M[i] - nM);
      M[i]     = nM;
      s0[i]    = __expf(s0[i] - nM);
      s1[i]    = __expf(s1[i] - nM);
      S[i]     = S[i] * corr[i] + rowsum16(s0[i] + s1[i]);
    }
#pragma unroll
    for (int d = 0; d < 16; ++d)
#pragma unroll
      for (int i = 0; i < 8; ++i) O[d][i] *= corr[i];

    // ---- P tile: C/D layout -> bf16 LDS -> A layout ------------------------
#pragma unroll
    for (int i = 0; i < 8; ++i) {
      pP[(i + 8 * half) * PP_S + lm]      = (__bf16)s0[i];
      pP[(i + 8 * half) * PP_S + 16 + lm] = (__bf16)s1[i];
    }
    v16bf pA;
    {
      const __bf16* pr = pP + lm * PP_S;
      v8bf a0 = *(const v8bf*)(pr + 8 * half);
      v8bf a1 = *(const v8bf*)(pr + 16 + 8 * half);
#pragma unroll
      for (int i = 0; i < 8; ++i) { pA[i] = a0[i]; pA[8 + i] = a1[i]; }
    }

    // ---- O += P * V  (B fragments from transposed KV tile) -----------------
#pragma unroll
    for (int d = 0; d < 16; ++d)
      O[d] = wmma_bf16(pA, bfrag16(&kvT[(d * 16 + lm) * KVT_S + 8 * half]), O[d]);

    __syncthreads();
  }

  // ---- normalize and stage O (bf16) for the A-layout FC fragments ---------
  float inv[8];
#pragma unroll
  for (int i = 0; i < 8; ++i) inv[i] = 1.f / S[i];
#pragma unroll
  for (int d = 0; d < 16; ++d)
#pragma unroll
    for (int i = 0; i < 8; ++i) {
      O[d][i] *= inv[i];
      oB[(i + 8 * half) * OB_S + d * 16 + lm] = (__bf16)O[d][i];
    }

  // ---- FC: relu(concat(q, o, q-o, q*o) @ W^T + bias) ----------------------
  v8f acc[8];
#pragma unroll
  for (int t = 0; t < 8; ++t)
#pragma unroll
    for (int i = 0; i < 8; ++i) acc[t][i] = 0.f;

  const __bf16* orow = oB + lm * OB_S;
#pragma unroll 4
  for (int c = 0; c < 32; ++c) {
    const int seg = c >> 3, cc = c & 7;
    const int col0 = 32 * cc + 8 * half;
    v16bf a;
    if (seg == 0) {
      a = qA[cc];                                   // bf16(q), already in A layout
    } else {
      v8bf o0 = *(const v8bf*)(orow + col0);
      v8bf o1 = *(const v8bf*)(orow + col0 + 16);
      if (seg == 1) {
#pragma unroll
        for (int i = 0; i < 8; ++i) { a[i] = o0[i]; a[8 + i] = o1[i]; }
      } else {
        vf4 x0 = *(const vf4*)(qrow + col0);
        vf4 x1 = *(const vf4*)(qrow + col0 + 4);
        vf4 y0 = *(const vf4*)(qrow + col0 + 16);
        vf4 y1 = *(const vf4*)(qrow + col0 + 20);
        float qv[16] = {x0.x, x0.y, x0.z, x0.w, x1.x, x1.y, x1.z, x1.w,
                        y0.x, y0.y, y0.z, y0.w, y1.x, y1.y, y1.z, y1.w};
#pragma unroll
        for (int i = 0; i < 8; ++i) {
          float olo = (float)o0[i], ohi = (float)o1[i];
          if (seg == 2) { a[i] = (__bf16)(qv[i] - olo); a[8 + i] = (__bf16)(qv[8 + i] - ohi); }
          else          { a[i] = (__bf16)(qv[i] * olo); a[8 + i] = (__bf16)(qv[8 + i] * ohi); }
        }
      }
    }
#pragma unroll
    for (int t = 0; t < 8; ++t) {
      const float* wr = W + (size_t)(t * 16 + lm) * FDIM + c * 32 + 8 * half;
      acc[t] = wmma_bf16(a, cvt16(wr, wr + 16), acc[t]);
    }
  }

#pragma unroll
  for (int t = 0; t < 8; ++t) {
    const float bv = bias[t * 16 + lm];
#pragma unroll
    for (int i = 0; i < 8; ++i) {
      float v = acc[t][i] + bv;
      v = v > 0.f ? v : 0.f;
      ob[(size_t)(qbase + i + 8 * half) * HID + t * 16 + lm] = v;
    }
  }
}

// ---------------------------------------------------------------------------
extern "C" void kernel_launch(void* const* d_in, const int* in_sizes, int n_in,
                              void* d_out, int out_size, void* d_ws, size_t ws_size,
                              hipStream_t stream) {
  (void)in_sizes; (void)n_in; (void)out_size; (void)d_ws; (void)ws_size;
  const float* cs = (const float*)d_in[0];  // [B, Lc, D]
  const float* rs = (const float*)d_in[1];  // [B, Lr, D]
  const float* W1 = (const float*)d_in[2];  // [H, 4D]
  const float* b1 = (const float*)d_in[3];  // [H]
  const float* W2 = (const float*)d_in[4];  // [H, 4D]
  const float* b2 = (const float*)d_in[5];  // [H]

  const int B = 32, Lc = 2048, Lr = 512;
  float* cl = (float*)d_out;                              // [B, Lc, H]
  float* rl = cl + (size_t)B * Lc * HID;                  // [B, Lr, H]

  // cl: Q = context, K/V = response (row softmax over r == alpha)
  attn_match_fc<<<dim3(Lc / 64, B), 128, SMEM_BYTES, stream>>>(
      cs, rs, W1, b1, cl, Lc, Lr);
  // rl: Q = response, K/V = context (row softmax of e^T == beta over c)
  attn_match_fc<<<dim3(Lr / 64, B), 128, SMEM_BYTES, stream>>>(
      rs, cs, W2, b2, rl, Lr, Lc);
}
